// DifferentiableH264_62251255988891
// MI455X (gfx1250) — matrix operand order repaired and checked by reading, not compiled, not run on hardware
//
#include <hip/hip_runtime.h>
#include <math.h>

typedef __attribute__((ext_vector_type(16))) _Float16 v16h;
typedef __attribute__((ext_vector_type(8)))  float    v8f;

// JPEG base quantization table (QF=28 -> scale = 200-2*28 = 144 -> *144/50 = *2.88)
__device__ __constant__ float BASEQ[64] = {
    16.f, 11.f, 10.f, 16.f,  24.f,  40.f,  51.f,  61.f,
    12.f, 12.f, 14.f, 19.f,  26.f,  58.f,  60.f,  55.f,
    14.f, 13.f, 16.f, 24.f,  40.f,  57.f,  69.f,  56.f,
    14.f, 17.f, 22.f, 29.f,  51.f,  87.f,  80.f,  62.f,
    18.f, 22.f, 37.f, 56.f,  68.f, 109.f, 103.f,  77.f,
    24.f, 35.f, 55.f, 64.f,  81.f, 104.f, 113.f,  92.f,
    49.f, 64.f, 78.f, 87.f, 103.f, 121.f, 120.f, 101.f,
    72.f, 92.f, 95.f, 98.f, 112.f, 100.f, 103.f,  99.f };

static __device__ __forceinline__ float dct_basis(int k, int n) {
    if (k == 0) return 0.35355339059327373f;                       // sqrt(1/8)
    return 0.5f * __cosf(0.19634954084936207f * (float)(k * (2 * n + 1))); // sqrt(2/8)*cos(pi*k*(2n+1)/16)
}

// C/D-layout f32 result -> A-operand (v16h) of its transpose: lane-local converts only.
static __device__ __forceinline__ v16h pack_a(const v8f d) {
    v16h a = {};
#pragma unroll
    for (int k = 0; k < 8; ++k) a[k] = (_Float16)d[k];             // K=16..31 (pad) stay zero
    return a;
}

__global__ __launch_bounds__(256) void dct_h264_wmma_kernel(const float* __restrict__ x,
                                                            float* __restrict__ out)
{
    constexpr int H = 1080, W = 1920;
    constexpr int PLANE = H * W;                 // per-channel plane
    constexpr int TILES_W = W / 32;              // 60 strips of 4 blocks
    constexpr int TILES_H = H / 8;               // 135 block rows

    const int lane = threadIdx.x & 31;
    const int wave = threadIdx.x >> 5;
    const int tile = blockIdx.x * 8 + wave;      // grid sized exactly: no guard, EXEC stays all-ones

    const int batch = tile / (TILES_H * TILES_W);
    const int rem   = tile % (TILES_H * TILES_W);
    const int bh    = rem / TILES_W;
    const int bt    = rem % TILES_W;

    // 16x16 virtual tile = 4 independent 8x8 blocks (quadrants) from an 8x32 strip.
    const int m    = lane & 15;
    const int hi   = lane >> 4;
    const int rowb = m & 7;
    const int quad = ((m >> 3) << 1) | hi;       // which 8-col group of the strip
    const int row  = bh * 8 + rowb;
    const int col0 = bt * 32 + quad * 8;

    const size_t base = (size_t)batch * 3 * PLANE + (size_t)row * W + col0;
    const float* pB = x + base;                  // channel order in reference: b, g, r
    const float* pG = pB + PLANE;
    const float* pR = pB + 2 * PLANE;

    float cb[8], cg[8], cr[8], y0[8];
    {
        float4 v0 = *(const float4*)(pB); float4 v1 = *(const float4*)(pB + 4);
        cb[0]=v0.x; cb[1]=v0.y; cb[2]=v0.z; cb[3]=v0.w; cb[4]=v1.x; cb[5]=v1.y; cb[6]=v1.z; cb[7]=v1.w;
        float4 g0 = *(const float4*)(pG); float4 g1 = *(const float4*)(pG + 4);
        cg[0]=g0.x; cg[1]=g0.y; cg[2]=g0.z; cg[3]=g0.w; cg[4]=g1.x; cg[5]=g1.y; cg[6]=g1.z; cg[7]=g1.w;
        float4 r0 = *(const float4*)(pR); float4 r1 = *(const float4*)(pR + 4);
        cr[0]=r0.x; cr[1]=r0.y; cr[2]=r0.z; cr[3]=r0.w; cr[4]=r1.x; cr[5]=r1.y; cr[6]=r1.z; cr[7]=r1.w;
    }
#pragma unroll
    for (int k = 0; k < 8; ++k)
        y0[k] = 0.114f * cb[k] + 0.587f * cg[k] + 0.299f * cr[k];

    // B-operand constants: Bt = block-diag(B8,B8)^T, Bc = block-diag(B8,B8).
    // B layout (32x16, K=16..31 zero-padded): lane l holds column N=l%16, halves[k] = Mat[k+8*(l/16)][N] for k<8.
    v16h Bt = {}; v16h Bc = {};
    {
        const int N = m;
#pragma unroll
        for (int k = 0; k < 8; ++k) {
            Bt[k] = (_Float16)((hi == 0) ? ((N <  8) ? dct_basis(N, k)     : 0.0f)   // B16^T[k][N]   = B16[N][k]
                                         : ((N >= 8) ? dct_basis(N - 8, k) : 0.0f)); // B16^T[k+8][N] = B16[N][k+8]
            Bc[k] = (_Float16)((hi == 0) ? ((N <  8) ? dct_basis(k, N)     : 0.0f)   // B16[k][N]
                                         : ((N >= 8) ? dct_basis(k, N - 8) : 0.0f)); // B16[k+8][N]
        }
    }

    // Quant row for the transposed coefficient tile D2 = C^T: element (VGPR r, lane l)
    // is C[l%8 (block-local u)][r (block-local v)] -> q[lane&7][r].
    // Scaled entries are all >= 28.8, so the max(.,1) clamp is dead; precompute the
    // reciprocal once per lane with v_rcp_f32 instead of 8 IEEE divides per tile.
    float qv[8], qi[8];
    {
        const int qr = (lane & 7) * 8;
#pragma unroll
        for (int j = 0; j < 8; ++j) {
            qv[j] = BASEQ[qr + j] * 2.88f;
            qi[j] = __builtin_amdgcn_rcpf(qv[j] + 1e-8f);
        }
    }

    // A1 = X (luma tile), K=0..15 live, K=16..31 zero.
    v16h a1 = {};
#pragma unroll
    for (int k = 0; k < 8; ++k) a1[k] = (_Float16)y0[k];

    const v8f cz = {};
    // D1 = X * B16^T
    v8f d1 = __builtin_amdgcn_wmma_f32_16x16x32_f16(false, a1, false, Bt, (short)0, cz, false, false);
    // D2 = (B16 * X^T) * B16^T = C^T
    v8f d2 = __builtin_amdgcn_wmma_f32_16x16x32_f16(false, pack_a(d1), false, Bt, (short)0, cz, false, false);

    // Hard quantization (forward value of the straight-through estimator), RTE like jnp.round.
#pragma unroll
    for (int k = 0; k < 8; ++k)
        d2[k] = rintf(d2[k] * qi[k]) * qv[k];

    // D3 = C_q * B16
    v8f d3 = __builtin_amdgcn_wmma_f32_16x16x32_f16(false, pack_a(d2), false, Bc, (short)0, cz, false, false);
    // D4 = (B16^T * C_q^T) * B16 = R^T  -> lands exactly on the positions this lane loaded.
    v8f d4 = __builtin_amdgcn_wmma_f32_16x16x32_f16(false, pack_a(d3), false, Bc, (short)0, cz, false, false);

    float ob[8], og[8], orr[8];
#pragma unroll
    for (int k = 0; k < 8; ++k) {
        const float yd = d4[k] - y0[k];
        ob[k]  = fminf(fmaxf(cb[k] + 0.114f * yd, 0.0f), 255.0f);
        og[k]  = fminf(fmaxf(cg[k] + 0.587f * yd, 0.0f), 255.0f);
        orr[k] = fminf(fmaxf(cr[k] + 0.299f * yd, 0.0f), 255.0f);
    }

    float* oB = out + base;
    float* oG = oB + PLANE;
    float* oR = oB + 2 * PLANE;
    *(float4*)(oB)     = make_float4(ob[0],  ob[1],  ob[2],  ob[3]);
    *(float4*)(oB + 4) = make_float4(ob[4],  ob[5],  ob[6],  ob[7]);
    *(float4*)(oG)     = make_float4(og[0],  og[1],  og[2],  og[3]);
    *(float4*)(oG + 4) = make_float4(og[4],  og[5],  og[6],  og[7]);
    *(float4*)(oR)     = make_float4(orr[0], orr[1], orr[2], orr[3]);
    *(float4*)(oR + 4) = make_float4(orr[4], orr[5], orr[6], orr[7]);
}

extern "C" void kernel_launch(void* const* d_in, const int* in_sizes, int n_in,
                              void* d_out, int out_size, void* d_ws, size_t ws_size,
                              hipStream_t stream) {
    (void)in_sizes; (void)n_in; (void)out_size; (void)d_ws; (void)ws_size;
    const float* x = (const float*)d_in[0];
    float* out = (float*)d_out;
    // 8 batches * 135 block-rows * 60 strips = 64800 tiles, 8 waves (tiles) per 256-thread block.
    const int blocks = (8 * 135 * 60) / 8;   // 8100, exact
    dct_h264_wmma_kernel<<<dim3(blocks), dim3(256), 0, stream>>>(x, out);
}
